// AutoCorrelation_56246891708841
// MI455X (gfx1250) — matrix operand order repaired and checked by reading, not compiled, not run on hardware
//
#include <hip/hip_runtime.h>
#include <math.h>

// ---------------------------------------------------------------------------
// AutoCorrelation attention for MI455X (gfx1250, wave32, WMMA + TDM).
// corr.mean(-1) == diagonal sums of Q K^T per head  =>  pure WMMA GEMMs, no FFT.
// ---------------------------------------------------------------------------

typedef __attribute__((ext_vector_type(16))) __bf16 v16bf;
typedef __attribute__((ext_vector_type(8)))  float  v8f;
typedef unsigned short u16t;
typedef __attribute__((ext_vector_type(8))) u16t u16x8;

constexpr int B_ = 8, L_ = 4096, DM = 512, H_ = 8, D_ = 64, KTOP = 8;
constexpr int M_ROWS = B_ * L_;                 // 32768

// ------------------- CDNA5 data-mover feature detection --------------------
#if __has_builtin(__builtin_amdgcn_tensor_load_to_lds) && \
    __has_builtin(__builtin_amdgcn_s_wait_tensorcnt)
  #define USE_TDM 1
#else
  #define USE_TDM 0
#endif
#if !USE_TDM && __has_builtin(__builtin_amdgcn_global_load_async_to_lds_b128)
  #define USE_ASYNC 1
#else
  #define USE_ASYNC 0
#endif

typedef __attribute__((address_space(1))) u16t gl_u16;   // global
typedef __attribute__((address_space(3))) u16t lo_u16;   // LDS
typedef __attribute__((address_space(3))) unsigned char lds_byte;

static __device__ __forceinline__ unsigned lds_offset(void* p) {
  return (unsigned)(unsigned long long)(lds_byte*)p;     // generic -> LDS byte addr
}

#if USE_TDM
typedef __attribute__((ext_vector_type(4))) unsigned int uint4v;
typedef __attribute__((ext_vector_type(8))) int          int8v;
typedef __attribute__((ext_vector_type(4))) int          int4v;

// DMA one 128-row x 32-col bf16 tile (row stride 512 elems) into LDS.
// D# layout per CDNA5 ISA ch.8: group0 = {flags, lds_addr, global_addr, type},
// group1 = {mask/data_size/pads, dims/tiles/strides}.
static __device__ __forceinline__ void tdm_tile_128x32(const u16t* gtile,
                                                       unsigned lds_addr,
                                                       unsigned tensor_rows) {
  unsigned long long ga = (unsigned long long)gtile;
  uint4v g0;
  g0[0] = 1u;                                            // count=1, no gather
  g0[1] = lds_addr;                                      // bits[63:32]
  g0[2] = (unsigned)(ga & 0xFFFFFFFFu);                  // global_addr lo
  g0[3] = (unsigned)((ga >> 32) & 0x01FFFFFFu) | (2u << 30);   // addr hi | type=2
  int8v g1;
  g1[0] = (int)(1u << 16);                               // data_size=1 (2 bytes)
  g1[1] = (int)((512u & 0xFFFFu) << 16);                 // tensor_dim0 lo16 @bits[63:48]
  g1[2] = (int)((512u >> 16) | ((tensor_rows & 0xFFFFu) << 16)); // dim0 hi | dim1 lo
  g1[3] = (int)((tensor_rows >> 16) | (32u << 16));      // dim1 hi | tile_dim0=32
  g1[4] = (int)128;                                      // tile_dim1=128, tile_dim2=0
  g1[5] = (int)512;                                      // tensor_dim0_stride lo32
  g1[6] = 0;                                             // stride0 hi | stride1 lo
  g1[7] = 0;
  int4v gz = {0, 0, 0, 0};
#if __clang_major__ >= 23
  int8v gz8 = {0, 0, 0, 0, 0, 0, 0, 0};
  __builtin_amdgcn_tensor_load_to_lds(g0, g1, gz, gz, gz8, 0);
#else
  __builtin_amdgcn_tensor_load_to_lds(g0, g1, gz, gz, 0);
#endif
}
#endif

static __device__ __forceinline__ u16t f32_to_bf16(float f) {
  union { float fv; unsigned uv; } x; x.fv = f;
  unsigned r = x.uv + 0x7FFFu + ((x.uv >> 16) & 1u);   // round-to-nearest-even
  return (u16t)(r >> 16);
}
static __device__ __forceinline__ float bf16_to_f32(u16t h) {
  union { unsigned uv; float fv; } x; x.uv = ((unsigned)h) << 16;
  return x.fv;
}
static __device__ __forceinline__ v16bf pack16(u16x8 lo, u16x8 hi) {
  union { u16x8 h[2]; v16bf v; } u;
  u.h[0] = lo; u.h[1] = hi;
  return u.v;
}

// --------------------------- f32 -> bf16 convert ---------------------------
__global__ void k_cvt_bf16(const float* __restrict__ in, u16t* __restrict__ out, int n) {
  int i = blockIdx.x * blockDim.x + threadIdx.x;
  if (i < n) out[i] = f32_to_bf16(in[i]);
}

// ----------------------- shared WMMA compute step --------------------------
// One 32-wide K step: 2 A frags + 4 B frags (ISA 16-bit layouts) -> 8 WMMAs.
static __device__ __forceinline__ void gemm_step(const u16t* sAbuf, const u16t* sWbuf,
                                                 int wm, int wn, int ln16, int lh,
                                                 v8f acc[2][4]) {
  v16bf afrag[2], bfrag[4];
#pragma unroll
  for (int ms = 0; ms < 2; ms++) {
    const int row = wm * 32 + ms * 16 + ln16;
    const u16t* p = &sAbuf[row * 32 + lh * 8];
    afrag[ms] = pack16(*(const u16x8*)p, *(const u16x8*)(p + 16));
  }
#pragma unroll
  for (int ns = 0; ns < 4; ns++) {
    const int nrow = wn * 64 + ns * 16 + ln16;
    const u16t* p = &sWbuf[nrow * 32 + lh * 8];
    bfrag[ns] = pack16(*(const u16x8*)p, *(const u16x8*)(p + 16));
  }
#pragma unroll
  for (int ms = 0; ms < 2; ms++)
#pragma unroll
    for (int ns = 0; ns < 4; ns++)
      acc[ms][ns] = __builtin_amdgcn_wmma_f32_16x16x32_bf16(
          false, afrag[ms], false, bfrag[ns], (short)0, acc[ms][ns], false, false);
}

// ------------------------------- WMMA GEMM ---------------------------------
// Y[m,n] = sum_k A[m,k]*W[n,k] + bias[n]   (torch Linear: x @ W.T + b)
// MODE 0: write bf16 head-major [B,H,L,D]   MODE 1: write f32 row-major [M,DM]
template <int MODE>
__global__ __launch_bounds__(256) void k_gemm(const u16t* __restrict__ A,
                                              const u16t* __restrict__ W,
                                              const float* __restrict__ bias,
                                              void* __restrict__ outp) {
  const int t    = threadIdx.x;
  const int wave = t >> 5, lane = t & 31;
  const int wm = wave & 3, wn = wave >> 2;     // 4 x 2 wave grid
  const int mBlk = blockIdx.x * 128;
  const int nBlk = blockIdx.y * 128;
  const int ln16 = lane & 15, lh = lane >> 4;

  v8f acc[2][4];
#pragma unroll
  for (int i = 0; i < 2; i++)
#pragma unroll
    for (int j = 0; j < 4; j++) acc[i][j] = {};

#if USE_TDM
  // -------- TDM double-buffered pipeline: DMA engine fills LDS ------------
  __shared__ u16t sA[2][128 * 32];
  __shared__ u16t sW[2][128 * 32];
  if (wave == 0) {
    tdm_tile_128x32(A + (size_t)mBlk * DM, lds_offset(&sA[0][0]), (unsigned)M_ROWS);
    tdm_tile_128x32(W + (size_t)nBlk * DM, lds_offset(&sW[0][0]), (unsigned)DM);
    __builtin_amdgcn_s_wait_tensorcnt(0);
  }
  __syncthreads();
  for (int kk = 0; kk < DM / 32; ++kk) {
    const int cur = kk & 1, nxt = cur ^ 1;
    if (wave == 0 && kk + 1 < DM / 32) {
      const int k0n = (kk + 1) * 32;
      tdm_tile_128x32(A + (size_t)mBlk * DM + k0n, lds_offset(&sA[nxt][0]), (unsigned)M_ROWS);
      tdm_tile_128x32(W + (size_t)nBlk * DM + k0n, lds_offset(&sW[nxt][0]), (unsigned)DM);
    }
    gemm_step(&sA[cur][0], &sW[cur][0], wm, wn, ln16, lh, acc);
    __syncthreads();                       // everyone done reading buf[cur]
    if (wave == 0) __builtin_amdgcn_s_wait_tensorcnt(0);
    __syncthreads();                       // buf[nxt] landed
  }
#elif USE_ASYNC
  // -------- ASYNCcnt double-buffered pipeline: async loads to LDS ---------
  __shared__ u16t sA[2][128 * 32];
  __shared__ u16t sW[2][128 * 32];
  const int row = t >> 1;
  const int half = (t & 1) * 16;
  auto stage = [&](int k0, int buf) {
    const u16t* gA = A + (size_t)(mBlk + row) * DM + k0 + half;
    const u16t* gW = W + (size_t)(nBlk + row) * DM + k0 + half;
    __builtin_amdgcn_global_load_async_to_lds_b128(
        (const gl_u16*)gA, (lo_u16*)&sA[buf][row * 32 + half], 0, 0);
    __builtin_amdgcn_global_load_async_to_lds_b128(
        (const gl_u16*)(gA + 8), (lo_u16*)&sA[buf][row * 32 + half + 8], 0, 0);
    __builtin_amdgcn_global_load_async_to_lds_b128(
        (const gl_u16*)gW, (lo_u16*)&sW[buf][row * 32 + half], 0, 0);
    __builtin_amdgcn_global_load_async_to_lds_b128(
        (const gl_u16*)(gW + 8), (lo_u16*)&sW[buf][row * 32 + half + 8], 0, 0);
  };
  stage(0, 0);
#if __has_builtin(__builtin_amdgcn_s_wait_asynccnt)
  __builtin_amdgcn_s_wait_asynccnt(0);
#else
  asm volatile("s_wait_asynccnt 0" ::: "memory");
#endif
  __syncthreads();
  for (int kk = 0; kk < DM / 32; ++kk) {
    const int cur = kk & 1, nxt = cur ^ 1;
    if (kk + 1 < DM / 32) stage((kk + 1) * 32, nxt);
    gemm_step(&sA[cur][0], &sW[cur][0], wm, wn, ln16, lh, acc);
    __syncthreads();
#if __has_builtin(__builtin_amdgcn_s_wait_asynccnt)
    __builtin_amdgcn_s_wait_asynccnt(0);
#else
    asm volatile("s_wait_asynccnt 0" ::: "memory");
#endif
    __syncthreads();
  }
#else
  // ------------------ fallback: synchronous staging -----------------------
  __shared__ u16t sA[128 * 32];
  __shared__ u16t sW[128 * 32];
  for (int k0 = 0; k0 < DM; k0 += 32) {
    {
      const int row = t >> 1;
      const int half = (t & 1) * 16;
      const u16t* gA = A + (size_t)(mBlk + row) * DM + k0 + half;
      *(u16x8*)&sA[row * 32 + half]     = *(const u16x8*)gA;
      *(u16x8*)&sA[row * 32 + half + 8] = *(const u16x8*)(gA + 8);
      const u16t* gW = W + (size_t)(nBlk + row) * DM + k0 + half;
      *(u16x8*)&sW[row * 32 + half]     = *(const u16x8*)gW;
      *(u16x8*)&sW[row * 32 + half + 8] = *(const u16x8*)(gW + 8);
    }
    __syncthreads();
    gemm_step(sA, sW, wm, wn, ln16, lh, acc);
    __syncthreads();
  }
#endif

  // epilogue: bias + store
#pragma unroll
  for (int ms = 0; ms < 2; ms++) {
#pragma unroll
    for (int ns = 0; ns < 4; ns++) {
      const int n = nBlk + wn * 64 + ns * 16 + ln16;
      const float bv = bias[n];
#pragma unroll
      for (int r = 0; r < 8; r++) {
        const int m = mBlk + wm * 32 + ms * 16 + r + (lh ? 8 : 0);
        const float val = acc[ms][ns][r] + bv;
        if (MODE == 0) {
          const int b = m >> 12, l = m & (L_ - 1);
          const int h = n >> 6, d = n & 63;
          ((u16t*)outp)[(((size_t)(b * H_ + h) * L_) + l) * D_ + d] = f32_to_bf16(val);
        } else {
          ((float*)outp)[(size_t)m * DM + n] = val;
        }
      }
    }
  }
}

// ------------------------- corr_mean via QK^T diagonals --------------------
__global__ __launch_bounds__(256) void k_corr(const u16t* __restrict__ qb,
                                              const u16t* __restrict__ kb,
                                              float* __restrict__ corr) {
  __shared__ float lag[L_];
  const int t    = threadIdx.x;
  const int wave = t >> 5, lane = t & 31;
  const int bh    = blockIdx.x >> 5;          // 32 j-strips per head
  const int jblk  = blockIdx.x & 31;
  const int jBase = jblk * 128 + wave * 16;
  const u16t* qh = qb + (size_t)bh * L_ * D_;
  const u16t* kh = kb + (size_t)bh * L_ * D_;
  const int ln16 = lane & 15, lh = lane >> 4;

  for (int j = t; j < L_; j += 256) lag[j] = 0.0f;

  // B fragments: k rows jBase..jBase+15, K=0..63 (resident across i loop)
  v16bf b0, b1;
  {
    const u16t* p = kh + (size_t)(jBase + ln16) * D_ + lh * 8;
    b0 = pack16(*(const u16x8*)p,        *(const u16x8*)(p + 16));
    b1 = pack16(*(const u16x8*)(p + 32), *(const u16x8*)(p + 48));
  }
  __syncthreads();

  const int jg = jBase + ln16;
  for (int it = 0; it < L_ / 16; ++it) {
    const int iBase = it * 16;
    const u16t* p = qh + (size_t)(iBase + ln16) * D_ + lh * 8;
    v16bf a0 = pack16(*(const u16x8*)p,        *(const u16x8*)(p + 16));
    v16bf a1 = pack16(*(const u16x8*)(p + 32), *(const u16x8*)(p + 48));
    __builtin_prefetch(p + 16 * D_, 0, 3);     // global_prefetch, keep near

    v8f c = {};
    c = __builtin_amdgcn_wmma_f32_16x16x32_bf16(false, a0, false, b0, (short)0, c, false, false);
    c = __builtin_amdgcn_wmma_f32_16x16x32_bf16(false, a1, false, b1, (short)0, c, false, false);

#pragma unroll
    for (int r = 0; r < 8; r++) {
      const int ig = iBase + r + (lh ? 8 : 0);
      const int d  = (ig - jg) & (L_ - 1);
      atomicAdd(&lag[d], c[r]);               // ds_add_f32
    }
  }
  __syncthreads();

  const float inv = 1.0f / (float)D_;
  float* ch = corr + (size_t)bh * L_;
  for (int j = t; j < L_; j += 256) atomicAdd(&ch[j], lag[j] * inv);
}

// ----------------------------- top-8 + softmax -----------------------------
__global__ __launch_bounds__(128) void k_topk(const float* __restrict__ corr,
                                              float* __restrict__ wts,
                                              int* __restrict__ idxs) {
  __shared__ float sv[128 * KTOP];
  __shared__ int   si[128 * KTOP];
  const int bh = blockIdx.x;
  const int t  = threadIdx.x;
  const float* c = corr + (size_t)bh * L_;

  float tv[KTOP]; int ti[KTOP];
#pragma unroll
  for (int k = 0; k < KTOP; k++) { tv[k] = -3.0e38f; ti[k] = 0; }
  for (int j = t; j < L_; j += 128) {
    const float v = c[j];
    if (v > tv[KTOP - 1]) {
      int k = KTOP - 1;
      while (k > 0 && tv[k - 1] < v) { tv[k] = tv[k - 1]; ti[k] = ti[k - 1]; --k; }
      tv[k] = v; ti[k] = j;
    }
  }
#pragma unroll
  for (int k = 0; k < KTOP; k++) { sv[t * KTOP + k] = tv[k]; si[t * KTOP + k] = ti[k]; }
  __syncthreads();

  if (t == 0) {
    float bv[KTOP]; int bi[KTOP];
    for (int k = 0; k < KTOP; k++) {
      float best = -3.0e38f; int bestp = 0;
      for (int p = 0; p < 128 * KTOP; p++)
        if (sv[p] > best) { best = sv[p]; bestp = p; }
      bv[k] = best; bi[k] = si[bestp]; sv[bestp] = -3.0e38f;
    }
    const float mx = bv[0];
    float e[KTOP], s = 0.f;
    for (int k = 0; k < KTOP; k++) { e[k] = __expf(bv[k] - mx); s += e[k]; }
    for (int k = 0; k < KTOP; k++) {
      wts[bh * KTOP + k]  = e[k] / s;
      idxs[bh * KTOP + k] = bi[k];
    }
  }
}

// -------------------- weighted rolled-V gather -> ctx ----------------------
__global__ __launch_bounds__(256) void k_gather(const u16t* __restrict__ vb,
                                                const float* __restrict__ wts,
                                                const int* __restrict__ idxs,
                                                u16t* __restrict__ ctx) {
  const int blk   = blockIdx.x;
  const int bh    = blk >> 7;                 // 128 blocks per head (32 l each)
  const int lBase = (blk & 127) * 32;
  const int t = threadIdx.x;
  const int l  = lBase + (t >> 3);
  const int dg = (t & 7) * 8;

  __shared__ float sw[KTOP];
  __shared__ int   sidx[KTOP];
  if (t < KTOP) { sw[t] = wts[bh * KTOP + t]; sidx[t] = idxs[bh * KTOP + t]; }
  __syncthreads();

  const u16t* vh = vb + (size_t)bh * L_ * D_;
  float acc[8];
#pragma unroll
  for (int i = 0; i < 8; i++) acc[i] = 0.f;
#pragma unroll
  for (int k = 0; k < KTOP; k++) {
    const int lp = (l + sidx[k]) & (L_ - 1);
    const u16x8 vv = *(const u16x8*)(vh + (size_t)lp * D_ + dg);
    const float w = sw[k];
#pragma unroll
    for (int i = 0; i < 8; i++) acc[i] += w * bf16_to_f32(vv[i]);
  }
  const int b = bh >> 3, h = bh & 7;
  u16x8 o;
#pragma unroll
  for (int i = 0; i < 8; i++) o[i] = f32_to_bf16(acc[i]);
  *(u16x8*)(ctx + ((size_t)(b * L_ + l) * DM) + h * D_ + dg) = o;
}

// ------------------------------- launcher ----------------------------------
extern "C" void kernel_launch(void* const* d_in, const int* in_sizes, int n_in,
                              void* d_out, int out_size, void* d_ws, size_t ws_size,
                              hipStream_t stream) {
  (void)in_sizes; (void)n_in; (void)out_size; (void)ws_size;
  const float* x  = (const float*)d_in[0];
  const float* Wq = (const float*)d_in[1];
  const float* bq = (const float*)d_in[2];
  const float* Wk = (const float*)d_in[3];
  const float* bk = (const float*)d_in[4];
  const float* Wv = (const float*)d_in[5];
  const float* bv = (const float*)d_in[6];
  const float* Wo = (const float*)d_in[7];
  const float* bo = (const float*)d_in[8];

  char* ws = (char*)d_ws;
  size_t off = 0;
  auto alloc = [&](size_t bytes) -> void* {
    void* p = ws + off;
    off += (bytes + 255) & ~(size_t)255;
    return p;
  };
  u16t* xb   = (u16t*)alloc((size_t)M_ROWS * DM * 2);
  u16t* Wqb  = (u16t*)alloc((size_t)DM * DM * 2);
  u16t* Wkb  = (u16t*)alloc((size_t)DM * DM * 2);
  u16t* Wvb  = (u16t*)alloc((size_t)DM * DM * 2);
  u16t* Wob  = (u16t*)alloc((size_t)DM * DM * 2);
  u16t* qb   = (u16t*)alloc((size_t)M_ROWS * DM * 2);
  u16t* kbuf = (u16t*)alloc((size_t)M_ROWS * DM * 2);
  u16t* vbuf = (u16t*)alloc((size_t)M_ROWS * DM * 2);
  float* corr = (float*)alloc((size_t)B_ * H_ * L_ * 4);
  float* wts  = (float*)alloc((size_t)B_ * H_ * KTOP * 4);
  int*   idxs = (int*)alloc((size_t)B_ * H_ * KTOP * 4);
  u16t* ctx   = (u16t*)alloc((size_t)M_ROWS * DM * 2);

  const int n_x = M_ROWS * DM;
  const int n_w = DM * DM;
  k_cvt_bf16<<<(n_x + 255) / 256, 256, 0, stream>>>(x, xb, n_x);
  k_cvt_bf16<<<(n_w + 255) / 256, 256, 0, stream>>>(Wq, Wqb, n_w);
  k_cvt_bf16<<<(n_w + 255) / 256, 256, 0, stream>>>(Wk, Wkb, n_w);
  k_cvt_bf16<<<(n_w + 255) / 256, 256, 0, stream>>>(Wv, Wvb, n_w);
  k_cvt_bf16<<<(n_w + 255) / 256, 256, 0, stream>>>(Wo, Wob, n_w);

  dim3 gg(M_ROWS / 128, DM / 128);
  k_gemm<0><<<gg, 256, 0, stream>>>(xb, Wqb, bq, qb);
  k_gemm<0><<<gg, 256, 0, stream>>>(xb, Wkb, bk, kbuf);
  k_gemm<0><<<gg, 256, 0, stream>>>(xb, Wvb, bv, vbuf);

  hipMemsetAsync(corr, 0, (size_t)B_ * H_ * L_ * 4, stream);
  k_corr<<<B_ * H_ * 32, 256, 0, stream>>>(qb, kbuf, corr);
  k_topk<<<B_ * H_, 128, 0, stream>>>(corr, wts, idxs);
  k_gather<<<B_ * H_ * (L_ / 32), 256, 0, stream>>>(vbuf, wts, idxs, ctx);

  k_gemm<1><<<gg, 256, 0, stream>>>(ctx, Wob, bo, d_out);
}
